// MyGNN_1443109011543
// MI455X (gfx1250) — compile-verified, matrix operand order checked by reading, not run on hardware
//
#include <hip/hip_runtime.h>
#include <hip/hip_fp16.h>

// MI455X / gfx1250, wave32. Fused GNN-ish op:
//   (1) pairwise exp-kernel reduction -> wsum, dwsum   (VALU + v_exp_f32, LDS-tiled)
//   (2) h = [x | x*(wsum*m-1) | dwsum*m] ; out = (h @ A + b) * m   (v_wmma_f32_16x16x32_f16)

typedef __attribute__((ext_vector_type(16))) _Float16 v16h;
typedef __attribute__((ext_vector_type(8)))  _Float16 v8h;
typedef __attribute__((ext_vector_type(8)))  float    v8f;
typedef __attribute__((ext_vector_type(4)))  int      v4i;

#define B_  32
#define N_  1024
#define F_  64
#define O_  128
#define K_  129            // 2*F + 1
#define KP_ 160            // K padded to 5 chunks of 32 for 16x16x32 WMMA
#define HS_ 168            // LDS row stride in halves (84 dwords -> conflict-free b128 rows)
#define M_  (B_ * N_)      // 32768 rows

#if __has_builtin(__builtin_amdgcn_global_load_async_to_lds_b128) && \
    __has_builtin(__builtin_amdgcn_s_wait_asynccnt)
#define USE_ASYNC_LDS 1
// From hipcc's diagnostic: param0 is "int4 __device__*" (addrspace(1)),
// so param1 should be the LDS-side "int4 __shared__*" (addrspace(3)).
typedef __attribute__((address_space(1))) v4i* gv4ip;
typedef __attribute__((address_space(3))) v4i* lv4ip;
#else
#define USE_ASYNC_LDS 0
#endif

// ---------------------------------------------------------------------------
// Kernel 0: weights A (129 x 128 f32, row-major) -> TRANSPOSED padded f16
// Aht[o][k], o in [0,128), k in [0,160) with zero pad for k >= 129.
// Transposed layout makes every WMMA B-fragment two contiguous b128 LDS loads.
// ---------------------------------------------------------------------------
__global__ void cvt_weights_kernel(const float* __restrict__ A, _Float16* __restrict__ Aht) {
    int idx = blockIdx.x * blockDim.x + threadIdx.x;
    if (idx >= KP_ * O_) return;
    int o = idx / KP_;
    int k = idx - o * KP_;
    float v = (k < K_) ? A[k * O_ + o] : 0.0f;
    Aht[idx] = (_Float16)v;
}

// ---------------------------------------------------------------------------
// Kernel 1: per-batch all-pairs Gaussian kernel sums.
// grid = B * (N/256) blocks of 256 threads; each thread owns one node j.
// All 1024 coords of the batch staged in LDS (8 KB); inner loop broadcasts
// coord i to all lanes (conflict-free) -> 2 subs, fma chain, v_exp_f32.
// ---------------------------------------------------------------------------
__global__ void pairwise_kernel(const float* __restrict__ x,
                                float* __restrict__ wsum,
                                float* __restrict__ dwsum) {
    __shared__ float2 c[N_];
    int batch = blockIdx.x >> 2;          // 4 blocks per batch
    int jbase = (blockIdx.x & 3) << 8;
    const float* xb = x + (size_t)batch * N_ * F_;

    for (int t = threadIdx.x; t < N_; t += 256) {
        c[t] = make_float2(xb[t * F_ + (F_ - 2)], xb[t * F_ + (F_ - 1)]);
    }
    __syncthreads();

    int j = jbase + threadIdx.x;
    float xj = c[j].x, yj = c[j].y;
    float ws = 0.0f, dw = 0.0f;
    #pragma unroll 4
    for (int i = 0; i < N_; ++i) {
        float dx = c[i].x - xj;
        float dy = c[i].y - yj;
        float d  = dx * dx + dy * dy;
        float w  = __expf(-10.0f * d);    // includes i==j (w=1), matching reference
        ws += w;
        dw += d * w;
    }
    wsum [batch * N_ + j] = ws;
    dwsum[batch * N_ + j] = dw;
}

// ---------------------------------------------------------------------------
// Kernel 2: fused h-build + GEMM (WMMA f16 -> f32 acc) + bias/mask epilogue.
// 256 threads = 8 waves per block; block owns 16 rows; wave w owns output
// columns [16w, 16w+16). K=160 -> 5 chunks of v_wmma_f32_16x16x32_f16.
// LDS: hTile 16 x HS_ f16 (5.25 KB) + weightsT 128 x HS_ f16 (42 KB).
// Fragments: ISA 7.12.2 16-bit packing -> per lane two contiguous 8-half
// runs per chunk -> 2x ds_load_b128 each for A and B.
// ---------------------------------------------------------------------------
__global__ void gemm_wmma_kernel(const float* __restrict__ x,
                                 const float* __restrict__ mask,
                                 const _Float16* __restrict__ Aht,
                                 const float* __restrict__ bias,
                                 const float* __restrict__ wsum,
                                 const float* __restrict__ dwsum,
                                 float* __restrict__ out) {
    __shared__ _Float16 hT [16 * HS_];    // h rows, [r][k], padded stride
    __shared__ _Float16 wTt[O_ * HS_];    // weights transposed, [o][k], padded stride

    int tid = threadIdx.x;
    int rowBase = blockIdx.x * 16;

    // ---- stage transposed weights (hot in L2 across all 2048 blocks) ----
#if USE_ASYNC_LDS
    {
        gv4ip gsrc = (gv4ip)(void*)Aht;           // [o][k]: 20 x 16B chunks per row
        lv4ip ldst = (lv4ip)(void*)wTt;           // [o][k] padded: 21 x 16B chunks per row
        for (int e = tid; e < O_ * 20; e += 256) {
            int o = e / 20;
            int j = e - o * 20;
            __builtin_amdgcn_global_load_async_to_lds_b128(
                gsrc + o * 20 + j, ldst + o * 21 + j, 0, 0);
        }
        __builtin_amdgcn_s_wait_asynccnt(0);
    }
#else
    for (int e = tid; e < O_ * KP_; e += 256) {
        int o = e / KP_;
        int k = e - o * KP_;
        wTt[o * HS_ + k] = Aht[e];
    }
#endif

    // ---- build h tile: h[r][k] = { x[k] ; x[k-64]*(wsum*m-1) ; dwsum*m ; 0 } ----
    for (int e = tid; e < 16 * KP_; e += 256) {
        int r = e / KP_;
        int k = e - r * KP_;
        int gr = rowBase + r;
        float v;
        if (k < F_) {
            v = x[(size_t)gr * F_ + k];
        } else if (k < 2 * F_) {
            float m = mask[gr];
            v = x[(size_t)gr * F_ + (k - F_)] * (wsum[gr] * m - 1.0f);
        } else if (k == 2 * F_) {
            v = dwsum[gr] * mask[gr];
        } else {
            v = 0.0f;
        }
        hT[r * HS_ + k] = (_Float16)v;
    }
    __syncthreads();

    int lane  = tid & 31;
    int wave  = tid >> 5;
    int rsel  = lane & 15;                // A: row index / B: column index
    int khalf = (lane >> 4) * 8;          // K sub-offset for upper half-wave

    const _Float16* arow = &hT [rsel * HS_ + khalf];
    const _Float16* brow = &wTt[(wave * 16 + rsel) * HS_ + khalf];

    v8f acc = {};
    #pragma unroll
    for (int c = 0; c < 5; ++c) {
        int kb = c * 32;
        v8h a0 = *(const v8h*)(arow + kb);        // k = kb+khalf   .. +7
        v8h a1 = *(const v8h*)(arow + kb + 16);   // k = kb+16+khalf.. +7
        v8h b0 = *(const v8h*)(brow + kb);
        v8h b1 = *(const v8h*)(brow + kb + 16);
        v16h af = __builtin_shufflevector(a0, a1, 0,1,2,3,4,5,6,7,8,9,10,11,12,13,14,15);
        v16h bf = __builtin_shufflevector(b0, b1, 0,1,2,3,4,5,6,7,8,9,10,11,12,13,14,15);
        acc = __builtin_amdgcn_wmma_f32_16x16x32_f16(
                  /*neg_a=*/false, af, /*neg_b=*/false, bf,
                  /*c_mod=*/(short)0, acc, /*reuse_a=*/false, /*reuse_b=*/false);
    }

    // D layout: vgpr r -> row r + 8*(lane/16), col = lane%16 (within wave's 16-col tile)
    int col  = wave * 16 + rsel;
    float bc = bias[col];
    #pragma unroll
    for (int r2 = 0; r2 < 8; ++r2) {
        int row = r2 + ((lane >> 4) << 3);
        int gr  = rowBase + row;
        out[(size_t)gr * O_ + col] = (acc[r2] + bc) * mask[gr];
    }
}

// ---------------------------------------------------------------------------
extern "C" void kernel_launch(void* const* d_in, const int* in_sizes, int n_in,
                              void* d_out, int out_size, void* d_ws, size_t ws_size,
                              hipStream_t stream) {
    const float* x    = (const float*)d_in[0];   // (32,1024,64)
    const float* mask = (const float*)d_in[1];   // (32,1024)
    const float* A    = (const float*)d_in[2];   // (129,128)
    const float* b    = (const float*)d_in[3];   // (128,)
    float* out = (float*)d_out;                  // (32,1024,128)

    char* ws = (char*)d_ws;
    float*    wsum  = (float*)(ws);                         // 32768 f32 = 128 KB
    float*    dwsum = (float*)(ws + (size_t)M_ * 4);        // 128 KB
    _Float16* Aht   = (_Float16*)(ws + (size_t)2 * M_ * 4); // 160*128 f16 = 40 KB (transposed)

    cvt_weights_kernel<<<(KP_ * O_ + 255) / 256, 256, 0, stream>>>(A, Aht);
    pairwise_kernel<<<B_ * (N_ / 256), 256, 0, stream>>>(x, wsum, dwsum);
    gemm_wmma_kernel<<<M_ / 16, 256, 0, stream>>>(x, mask, Aht, b, wsum, dwsum, out);
}
